// MambaEncoder_33500744908981
// MI455X (gfx1250) — matrix operand order repaired and checked by reading, not compiled, not run on hardware
//
#include <hip/hip_runtime.h>

// ---------------------------------------------------------------------------
// MambaEncoder for MI455X (gfx1250, wave32, WMMA).
// Dense GEMMs -> v_wmma_f32_16x16x32_f16, 32x64 wave tile, double-buffered
// fragment loads (no tail guards -> no EXEC masking around WMMA).
// Scan / softmax / conv / norm -> fp32 VALU with LDS staging + shfl reduce.
// ---------------------------------------------------------------------------

typedef __attribute__((ext_vector_type(16))) _Float16 v16h;
typedef __attribute__((ext_vector_type(8)))  _Float16 v8h;
typedef __attribute__((ext_vector_type(8)))  float    v8f;
typedef _Float16 half_t;

#define L_SEQ   2048
#define D_MODEL 1024
#define OUT_DIM 512
#define D_STATE 16
#define D_CONV  4
#define D_INNER 2048
#define DT_RANK 64
#define XP_PAD  128   // x_proj rows padded 96 -> 128 so N%64==0

// ---------------- workspace layout (bytes) ----------------------------------
constexpr size_t OFF_TOK_H  = 0;                                   // 2048x1024 f16
constexpr size_t OFF_INW_H  = OFF_TOK_H  + 2048ull*1024*2;         // 4096x1024 f16
constexpr size_t OFF_XPW_H  = OFF_INW_H  + 4096ull*1024*2;         // 128x2048  f16 (rows 96..127 zero)
constexpr size_t OFF_DTW_H  = OFF_XPW_H  + (size_t)XP_PAD*2048*2;  // 2048x64   f16
constexpr size_t OFF_OUTW_H = OFF_DTW_H  + 2048ull*64*2;           // 1024x2048 f16
constexpr size_t OFF_IMG_H  = OFF_OUTW_H + 1024ull*2048*2;         // 2048x512  f16
constexpr size_t OFF_TXT_H  = OFF_IMG_H  + 2048ull*512*2;          // 2048x512  f16
constexpr size_t OFF_TXTT_H = OFF_TXT_H  + 2048ull*512*2;          // 512x2048  f16
constexpr size_t OFF_XZ     = OFF_TXTT_H + 512ull*2048*2;          // 2048x4096 f32 (later: S 2048x2048 f32)
constexpr size_t OFF_XH     = OFF_XZ     + 2048ull*4096*4;         // 2048x2048 f16
constexpr size_t OFF_XDBL   = OFF_XH     + 2048ull*2048*2;         // 2048x128  f32
constexpr size_t OFF_DTRH   = OFF_XDBL   + 2048ull*XP_PAD*4;       // 2048x64   f16
constexpr size_t OFF_DT     = OFF_DTRH   + 2048ull*64*2;           // 2048x2048 f32 (later: P 2048x2048 f16)
constexpr size_t OFF_YH     = OFF_DT     + 2048ull*2048*4;         // 2048x2048 f16
constexpr size_t OFF_MAMBA  = OFF_YH     + 2048ull*2048*2;         // 2048x1024 f32
constexpr size_t OFF_CAM    = OFF_MAMBA  + 2048ull*1024*4;         // 2048x512  f32

// ---------------------------------------------------------------------------
// WMMA GEMM: C[M,N] = A[M,K] * B^T, A (M,K) row-major f16, B (N,K) row-major
// f16 (column-n of B^T is K-contiguous -> direct b128 fragment loads).
// Wave tile 32x64: 2 A-frags x 4 B-frags -> 8 WMMA per K-step, software
// pipelined one K-step deep. Requires M%32==0, N%64==0, K%32==0.
// ---------------------------------------------------------------------------
__device__ __forceinline__ v16h ld_frag_pair(const half_t* p0, const half_t* p1) {
    v8h a = *reinterpret_cast<const v8h*>(p0);
    v8h b = *reinterpret_cast<const v8h*>(p1);
    v16h r;
#pragma unroll
    for (int i = 0; i < 8; ++i) { r[i] = a[i]; r[i + 8] = b[i]; }
    return r;
}

#define WMMA_F16(Af, Bf, Cf) \
    __builtin_amdgcn_wmma_f32_16x16x32_f16(false, (Af), false, (Bf), (short)0, (Cf), false, false)

__launch_bounds__(256) __global__
void k_gemm_f16(const half_t* __restrict__ A, const half_t* __restrict__ B,
                float* __restrict__ C, int M, int N, int K,
                int lda, int ldb, int ldc) {
    const int nGroups = N >> 6;
    const int totalW  = (M >> 5) * nGroups;
    const int wid     = blockIdx.x * 8 + (threadIdx.x >> 5);
    if (wid >= totalW) return;

    const int mTile  = wid / nGroups;
    const int nGroup = wid - mTile * nGroups;
    const int m0 = mTile << 5;
    const int n0 = nGroup << 6;

    const int lane = threadIdx.x & 31;
    const int r    = lane & 15;     // row (A) / column (B) within 16
    const int hi   = lane >> 4;     // half-wave selector

    // A-frag: lane<16 holds K 0..15 of row r; lane>=16 holds K 16..31 of row r
    const half_t* a0 = A + (size_t)(m0 + r) * lda + hi * 16;
    const half_t* a1 = a0 + (size_t)16 * lda;
    // B-frag: lane<16 holds K 0..15 of col r; lane>=16 holds K 16..31 of col r
    const half_t* b0 = B + (size_t)(n0 + r) * ldb + hi * 16;
    const half_t* b1 = b0 + (size_t)16 * ldb;
    const half_t* b2 = b0 + (size_t)32 * ldb;
    const half_t* b3 = b0 + (size_t)48 * ldb;

    v8f c00 = {}, c01 = {}, c02 = {}, c03 = {};
    v8f c10 = {}, c11 = {}, c12 = {}, c13 = {};

    v16h A0 = ld_frag_pair(a0, a0 + 8);
    v16h A1 = ld_frag_pair(a1, a1 + 8);
    v16h B0 = ld_frag_pair(b0, b0 + 8);
    v16h B1 = ld_frag_pair(b1, b1 + 8);
    v16h B2 = ld_frag_pair(b2, b2 + 8);
    v16h B3 = ld_frag_pair(b3, b3 + 8);

    for (int k0 = 32; k0 < K; k0 += 32) {
        // issue next K-step's 12 fragment loads before this step's WMMAs
        v16h nA0 = ld_frag_pair(a0 + k0, a0 + k0 + 8);
        v16h nA1 = ld_frag_pair(a1 + k0, a1 + k0 + 8);
        v16h nB0 = ld_frag_pair(b0 + k0, b0 + k0 + 8);
        v16h nB1 = ld_frag_pair(b1 + k0, b1 + k0 + 8);
        v16h nB2 = ld_frag_pair(b2 + k0, b2 + k0 + 8);
        v16h nB3 = ld_frag_pair(b3 + k0, b3 + k0 + 8);
        // keep the tile after next warm in the near (WGP-level) caches
        __builtin_prefetch(a0 + k0 + 32, 0, 3);
        __builtin_prefetch(b0 + k0 + 32, 0, 3);

        c00 = WMMA_F16(A0, B0, c00);
        c01 = WMMA_F16(A0, B1, c01);
        c02 = WMMA_F16(A0, B2, c02);
        c03 = WMMA_F16(A0, B3, c03);
        c10 = WMMA_F16(A1, B0, c10);
        c11 = WMMA_F16(A1, B1, c11);
        c12 = WMMA_F16(A1, B2, c12);
        c13 = WMMA_F16(A1, B3, c13);

        A0 = nA0; A1 = nA1;
        B0 = nB0; B1 = nB1; B2 = nB2; B3 = nB3;
    }
    c00 = WMMA_F16(A0, B0, c00);
    c01 = WMMA_F16(A0, B1, c01);
    c02 = WMMA_F16(A0, B2, c02);
    c03 = WMMA_F16(A0, B3, c03);
    c10 = WMMA_F16(A1, B0, c10);
    c11 = WMMA_F16(A1, B1, c11);
    c12 = WMMA_F16(A1, B2, c12);
    c13 = WMMA_F16(A1, B3, c13);

    // C/D layout: VGPR i -> (M = m0 + i + 8*hi [+16], N = n0 + 16j + r)
    const int col0 = n0 + r;
#pragma unroll
    for (int i = 0; i < 8; ++i) {
        float* Crow  = C + (size_t)(m0 + i + hi * 8) * ldc;
        float* Crow2 = Crow + (size_t)16 * ldc;
        Crow[col0]       = c00[i];
        Crow[col0 + 16]  = c01[i];
        Crow[col0 + 32]  = c02[i];
        Crow[col0 + 48]  = c03[i];
        Crow2[col0]      = c10[i];
        Crow2[col0 + 16] = c11[i];
        Crow2[col0 + 32] = c12[i];
        Crow2[col0 + 48] = c13[i];
    }
}

// ---------------------------------------------------------------------------
// Elementwise / conversion kernels
// ---------------------------------------------------------------------------
__global__ void k_f32_to_f16(const float* __restrict__ src, half_t* __restrict__ dst, int n) {
    int i = blockIdx.x * 256 + threadIdx.x;
    if (i < n) dst[i] = (half_t)src[i];
}

__global__ void k_zero_f16(half_t* __restrict__ dst, int n) {
    int i = blockIdx.x * 256 + threadIdx.x;
    if (i < n) dst[i] = (half_t)0.0f;
}

__global__ void k_tokens(const float* __restrict__ img, const float* __restrict__ txt,
                         half_t* __restrict__ tok) {
    int idx = blockIdx.x * 256 + threadIdx.x;           // L*D_MODEL
    if (idx >= L_SEQ * D_MODEL) return;
    int t = idx >> 10, c = idx & 1023;
    float v = (c < OUT_DIM) ? img[(size_t)t * OUT_DIM + c]
                            : txt[(size_t)t * OUT_DIM + (c - OUT_DIM)];
    tok[idx] = (half_t)v;
}

__global__ void k_transpose_txt(const float* __restrict__ src, half_t* __restrict__ dst) {
    // src (L,OUT_DIM) f32 -> dst (OUT_DIM,L) f16
    int idx = blockIdx.x * 256 + threadIdx.x;           // OUT_DIM*L
    if (idx >= OUT_DIM * L_SEQ) return;
    int n = idx >> 11, k = idx & 2047;
    dst[idx] = (half_t)src[(size_t)k * OUT_DIM + n];
}

__global__ void k_conv_silu(const float* __restrict__ xz, const float* __restrict__ w,
                            const float* __restrict__ b, half_t* __restrict__ xh) {
    int idx = blockIdx.x * 256 + threadIdx.x;           // L*D_INNER
    if (idx >= L_SEQ * D_INNER) return;
    int t = idx >> 11, d = idx & 2047;
    float acc = b[d];
#pragma unroll
    for (int k = 0; k < D_CONV; ++k) {
        int tt = t + k - (D_CONV - 1);
        if (tt >= 0) acc = fmaf(w[d * D_CONV + k], xz[(size_t)tt * 4096 + d], acc);
    }
    float s = acc / (1.0f + __expf(-acc));              // silu
    xh[(size_t)t * D_INNER + d] = (half_t)s;
}

__global__ void k_extract_dtr(const float* __restrict__ xdbl, half_t* __restrict__ dtr) {
    int idx = blockIdx.x * 256 + threadIdx.x;           // L*DT_RANK
    if (idx >= L_SEQ * DT_RANK) return;
    int t = idx >> 6, c = idx & 63;
    dtr[idx] = (half_t)xdbl[(size_t)t * XP_PAD + c];
}

__global__ void k_softplus_bias(float* __restrict__ dt, const float* __restrict__ b) {
    int idx = blockIdx.x * 256 + threadIdx.x;           // L*D_INNER
    if (idx >= L_SEQ * D_INNER) return;
    int d = idx & 2047;
    float v = dt[idx] + b[d];
    dt[idx] = (v > 20.0f) ? v : log1pf(__expf(v));
}

// ---------------------------------------------------------------------------
// Selective scan: thread=(d,n), 16 channels x 16 states per 256-thread block.
// 64-timestep LDS staging; 16-lane shfl_xor reduce over the state dim;
// fused epilogue y = (ys + D*x) * silu(z) -> f16 for the out_proj GEMM.
// ---------------------------------------------------------------------------
#define TCH 64
__launch_bounds__(256) __global__
void k_scan(const float* __restrict__ xdbl, const float* __restrict__ dt,
            const half_t* __restrict__ xh, const float* __restrict__ xz,
            const float* __restrict__ A_log, const float* __restrict__ Dp,
            half_t* __restrict__ yh) {
    __shared__ float sB[TCH][D_STATE];
    __shared__ float sC[TCH][D_STATE];
    __shared__ float sdt[TCH][16];
    __shared__ float sx[TCH][16];

    const int tid = threadIdx.x;
    const int dL  = tid >> 4;                 // 0..15 channel-in-block
    const int n   = tid & 15;                 // state index
    const int d   = blockIdx.x * 16 + dL;

    const float Adn = -__expf(A_log[(size_t)d * D_STATE + n]);
    const float Dd  = Dp[d];
    float h = 0.0f;

    for (int t0 = 0; t0 < L_SEQ; t0 += TCH) {
        for (int i = tid; i < TCH * D_STATE; i += 256) {
            int tt = i >> 4, nn = i & 15, t = t0 + tt;
            sB[tt][nn] = xdbl[(size_t)t * XP_PAD + DT_RANK + nn];
            sC[tt][nn] = xdbl[(size_t)t * XP_PAD + DT_RANK + D_STATE + nn];
        }
        for (int i = tid; i < TCH * 16; i += 256) {
            int tt = i >> 4, dd = i & 15, t = t0 + tt;
            int dg = blockIdx.x * 16 + dd;
            sdt[tt][dd] = dt[(size_t)t * D_INNER + dg];
            sx[tt][dd]  = (float)xh[(size_t)t * D_INNER + dg];
        }
        __syncthreads();

        for (int tt = 0; tt < TCH; ++tt) {
            float dtv = sdt[tt][dL];
            float xv  = sx[tt][dL];
            float dA  = __expf(dtv * Adn);
            h = fmaf(dA, h, dtv * xv * sB[tt][n]);
            float v = h * sC[tt][n];
            v += __shfl_xor(v, 8, 16);
            v += __shfl_xor(v, 4, 16);
            v += __shfl_xor(v, 2, 16);
            v += __shfl_xor(v, 1, 16);
            if (n == 0) {
                int t = t0 + tt;
                float zv = xz[(size_t)t * 4096 + D_INNER + d];
                float y  = (v + Dd * xv) * (zv / (1.0f + __expf(-zv)));
                yh[(size_t)t * D_INNER + d] = (half_t)y;
            }
        }
        __syncthreads();
    }
}

// ---------------------------------------------------------------------------
// Row softmax (f32 scores -> f16 probabilities)
// ---------------------------------------------------------------------------
__launch_bounds__(256) __global__
void k_softmax_row(const float* __restrict__ S, half_t* __restrict__ P, int n) {
    __shared__ float red[256];
    const int row = blockIdx.x, tid = threadIdx.x;
    const float* s = S + (size_t)row * n;
    float m = -3.0e38f;
    for (int c = tid; c < n; c += 256) m = fmaxf(m, s[c]);
    red[tid] = m; __syncthreads();
    for (int o = 128; o > 0; o >>= 1) { if (tid < o) red[tid] = fmaxf(red[tid], red[tid + o]); __syncthreads(); }
    m = red[0]; __syncthreads();
    float sum = 0.0f;
    for (int c = tid; c < n; c += 256) sum += __expf(s[c] - m);
    red[tid] = sum; __syncthreads();
    for (int o = 128; o > 0; o >>= 1) { if (tid < o) red[tid] += red[tid + o]; __syncthreads(); }
    float inv = 1.0f / red[0];
    half_t* p = P + (size_t)row * n;
    for (int c = tid; c < n; c += 256) p[c] = (half_t)(__expf(s[c] - m) * inv);
}

// ---------------------------------------------------------------------------
// Finalize: res = mamba_out + [alpha*cam_out | image_embed]; L2 row-normalize;
// split into the two OUT_DIM-wide outputs concatenated flat.
// ---------------------------------------------------------------------------
__launch_bounds__(256) __global__
void k_finalize(const float* __restrict__ mamba, const float* __restrict__ cam,
                const float* __restrict__ img, const float* __restrict__ alpha,
                float* __restrict__ out) {
    __shared__ float red[256];
    const int row = blockIdx.x, tid = threadIdx.x;
    const float a = alpha[0];
    float v[4], ss = 0.0f;
#pragma unroll
    for (int j = 0; j < 4; ++j) {
        int c = tid + j * 256;
        float m = mamba[(size_t)row * D_MODEL + c];
        float add = (c < OUT_DIM) ? a * cam[(size_t)row * OUT_DIM + c]
                                  : img[(size_t)row * OUT_DIM + (c - OUT_DIM)];
        v[j] = m + add;
        ss = fmaf(v[j], v[j], ss);
    }
    red[tid] = ss; __syncthreads();
    for (int o = 128; o > 0; o >>= 1) { if (tid < o) red[tid] += red[tid + o]; __syncthreads(); }
    float inv = 1.0f / fmaxf(sqrtf(red[0]), 1e-12f);
#pragma unroll
    for (int j = 0; j < 4; ++j) {
        int c = tid + j * 256;
        float r = v[j] * inv;
        if (c < OUT_DIM) out[(size_t)row * OUT_DIM + c] = r;
        else out[(size_t)L_SEQ * OUT_DIM + (size_t)row * OUT_DIM + (c - OUT_DIM)] = r;
    }
}

// ---------------------------------------------------------------------------
// Launcher
// ---------------------------------------------------------------------------
extern "C" void kernel_launch(void* const* d_in, const int* in_sizes, int n_in,
                              void* d_out, int out_size, void* d_ws, size_t ws_size,
                              hipStream_t stream) {
    const float* img   = (const float*)d_in[0];
    const float* txt   = (const float*)d_in[1];
    const float* inW   = (const float*)d_in[2];
    const float* convW = (const float*)d_in[3];
    const float* convB = (const float*)d_in[4];
    const float* xpW   = (const float*)d_in[5];
    const float* dtW   = (const float*)d_in[6];
    const float* dtB   = (const float*)d_in[7];
    const float* A_log = (const float*)d_in[8];
    const float* Dp    = (const float*)d_in[9];
    const float* outW  = (const float*)d_in[10];
    const float* alpha = (const float*)d_in[11];
    float* out = (float*)d_out;

    char* w = (char*)d_ws;
    half_t* tokH  = (half_t*)(w + OFF_TOK_H);
    half_t* inWH  = (half_t*)(w + OFF_INW_H);
    half_t* xpWH  = (half_t*)(w + OFF_XPW_H);
    half_t* dtWH  = (half_t*)(w + OFF_DTW_H);
    half_t* outWH = (half_t*)(w + OFF_OUTW_H);
    half_t* imgH  = (half_t*)(w + OFF_IMG_H);
    half_t* txtH  = (half_t*)(w + OFF_TXT_H);
    half_t* txtTH = (half_t*)(w + OFF_TXTT_H);
    float*  xz    = (float*) (w + OFF_XZ);
    half_t* xh    = (half_t*)(w + OFF_XH);
    float*  xdbl  = (float*) (w + OFF_XDBL);
    half_t* dtrH  = (half_t*)(w + OFF_DTRH);
    float*  dtF   = (float*) (w + OFF_DT);
    half_t* yh    = (half_t*)(w + OFF_YH);
    float*  mamba = (float*) (w + OFF_MAMBA);
    float*  cam   = (float*) (w + OFF_CAM);
    float*  S     = xz;                         // alias: xz dead after scan
    half_t* P     = (half_t*)(w + OFF_DT);      // alias: dt dead after scan

    auto cvt = [&](const float* s, half_t* d, int n) {
        k_f32_to_f16<<<(n + 255) / 256, 256, 0, stream>>>(s, d, n);
    };
    auto gemm = [&](const half_t* A, const half_t* B, float* C,
                    int M, int N, int K, int lda, int ldb, int ldc) {
        int waves  = (M / 32) * (N / 64);
        int blocks = (waves + 7) / 8;
        k_gemm_f16<<<blocks, 256, 0, stream>>>(A, B, C, M, N, K, lda, ldb, ldc);
    };

    // 1) f16 conversions / layout prep (x_proj padded to 128 rows, tail zeroed)
    cvt(inW,  inWH,  4096 * 1024);
    cvt(xpW,  xpWH,  96 * 2048);
    k_zero_f16<<<(32 * 2048 + 255) / 256, 256, 0, stream>>>(xpWH + 96 * 2048, 32 * 2048);
    cvt(dtW,  dtWH,  2048 * 64);
    cvt(outW, outWH, 1024 * 2048);
    cvt(img,  imgH,  L_SEQ * OUT_DIM);
    cvt(txt,  txtH,  L_SEQ * OUT_DIM);
    k_tokens<<<(L_SEQ * D_MODEL + 255) / 256, 256, 0, stream>>>(img, txt, tokH);
    k_transpose_txt<<<(OUT_DIM * L_SEQ + 255) / 256, 256, 0, stream>>>(txt, txtTH);

    // 2) xz = tokens @ in_proj_w.T           (2048 x 4096, K=1024)
    gemm(tokH, inWH, xz, L_SEQ, 2 * D_INNER, D_MODEL, D_MODEL, D_MODEL, 2 * D_INNER);

    // 3) x = silu(causal depthwise conv(xz[:, :2048]))
    k_conv_silu<<<(L_SEQ * D_INNER + 255) / 256, 256, 0, stream>>>(xz, convW, convB, xh);

    // 4) x_dbl = x @ x_proj_w.T              (2048 x 128pad, K=2048)
    gemm(xh, xpWH, xdbl, L_SEQ, XP_PAD, D_INNER, D_INNER, D_INNER, XP_PAD);
    k_extract_dtr<<<(L_SEQ * DT_RANK + 255) / 256, 256, 0, stream>>>(xdbl, dtrH);

    // 5) dt = softplus(dt_r @ dt_proj_w.T + b)   (2048 x 2048, K=64)
    gemm(dtrH, dtWH, dtF, L_SEQ, D_INNER, DT_RANK, DT_RANK, DT_RANK, D_INNER);
    k_softplus_bias<<<(L_SEQ * D_INNER + 255) / 256, 256, 0, stream>>>(dtF, dtB);

    // 6) selective scan + gating epilogue -> yh (f16)
    k_scan<<<D_INNER / 16, 256, 0, stream>>>(xdbl, dtF, xh, xz, A_log, Dp, yh);

    // 7) mamba_out = y @ out_proj_w.T        (2048 x 1024, K=2048)
    gemm(yh, outWH, mamba, L_SEQ, D_MODEL, D_INNER, D_INNER, D_INNER, D_MODEL);

    // 8) attention: S = img @ txt.T ; P = softmax(S) ; cam = P @ txt
    gemm(imgH, txtH, S, L_SEQ, L_SEQ, OUT_DIM, OUT_DIM, OUT_DIM, L_SEQ);
    k_softmax_row<<<L_SEQ, 256, 0, stream>>>(S, P, L_SEQ);
    gemm(P, txtTH, cam, L_SEQ, OUT_DIM, L_SEQ, L_SEQ, L_SEQ, OUT_DIM);

    // 9) residual add + row L2 normalize + split outputs
    k_finalize<<<L_SEQ, 256, 0, stream>>>(mamba, cam, img, alpha, out);

    (void)in_sizes; (void)n_in; (void)out_size; (void)ws_size;
}
